// HANModel_55284819034410
// MI455X (gfx1250) — matrix operand order repaired and checked by reading, not compiled, not run on hardware
//
#include <hip/hip_runtime.h>
#include <hip/hip_bf16.h>

// ---------------------------------------------------------------------------
// HANConv (2 relations, 2 layers) on MI455X / gfx1250.
// Dense GEMMs  -> v_wmma_f32_16x16x32_bf16 (fp32 accumulate)
// Edge phases  -> gather + L2-resident f32 atomics (segment softmax/scatter)
// ---------------------------------------------------------------------------

#define DEVI __device__ __forceinline__

typedef __attribute__((ext_vector_type(16))) __bf16 v16bf;
typedef __attribute__((ext_vector_type(8)))  float  v8f;

union Frag16 { v16bf bf; unsigned u[8]; };

static constexpr int NN   = 200000;
static constexpr int EE   = 1000000;
static constexpr int HIDC = 96;

DEVI unsigned short f2bf(float f) {                 // round-to-nearest-even
  unsigned u = __float_as_uint(f);
  return (unsigned short)((u + 0x7FFFu + ((u >> 16) & 1u)) >> 16);
}
DEVI unsigned fkey(float f) {                       // order-preserving f32->u32
  unsigned u = __float_as_uint(f);
  return (u & 0x80000000u) ? ~u : (u | 0x80000000u);
}
DEVI float funkey(unsigned k) {
  return __uint_as_float((k & 0x80000000u) ? (k ^ 0x80000000u) : ~k);
}

// --------------------------- conversion / init ----------------------------
__global__ void k_cvt_bf16(const float* __restrict__ x,
                           unsigned short* __restrict__ o, int n) {
  int t = blockIdx.x * blockDim.x + threadIdx.x;
  if (t < n) o[t] = f2bf(x[t]);
}

__global__ void k_zero_f32(float* p, int n) {
  int t = blockIdx.x * blockDim.x + threadIdx.x;
  if (t < n) p[t] = 0.f;
}

// zero out[N*96]; mu/s[N*3]; mu=0 is the -inf key
__global__ void k_init_rel(unsigned* __restrict__ mu, float* __restrict__ s,
                           float* __restrict__ out, int n96, int n3) {
  int t = blockIdx.x * blockDim.x + threadIdx.x;
  if (t >= n96) return;
  out[t] = 0.f;
  if (t < n3) { mu[t] = 0u; s[t] = 0.f; }
}

// Pack W[K,96] fp32 into B-fragment-native bf16 pairs:
// idx = ((kk*6+nt)*32 + lane)*8 + v ; lane = column n (mod 16), hi lanes K+8; v>=4 holds K+16
__global__ void k_pack_w(const float* __restrict__ W,
                         unsigned* __restrict__ pack, int K) {
  int t = blockIdx.x * blockDim.x + threadIdx.x;
  int total = (K >> 5) * 6 * 32 * 8;
  if (t >= total) return;
  int v    = t & 7;
  int u    = t >> 3;
  int lane = u & 31;
  int w    = u >> 5;
  int nt   = w % 6;
  int kk   = w / 6;
  int n  = nt * 16 + (lane & 15);
  int k0 = kk * 32 + ((lane >= 16) ? 8 : 0) + ((v < 4) ? 2 * v : 16 + 2 * (v - 4));
  unsigned lo = f2bf(W[(k0    ) * HIDC + n]);
  unsigned hi = f2bf(W[(k0 + 1) * HIDC + n]);
  pack[t] = lo | (hi << 16);
}

// --------------------------- WMMA GEMM: C = A@W + b ------------------------
// A: [Nrows, K] bf16 (row major), Wpack: fragment-packed bf16, C: [Nrows,96]
__global__ void k_gemm(const unsigned* __restrict__ A32,
                       const unsigned* __restrict__ Wpack,
                       const float* __restrict__ bias,
                       float* __restrict__ C, int K, int total_tiles) {
  int wid = (blockIdx.x * blockDim.x + threadIdx.x) >> 5;   // wave-uniform
  if (wid >= total_tiles) return;
  int lane  = threadIdx.x & 31;
  int mtile = wid / 6, ntile = wid - mtile * 6;
  int row   = mtile * 16 + (lane & 15);
  int koff  = (lane >> 4) * 8;                 // A: hi-half lanes hold K+8
  int rowK2 = (row * K) >> 1;                  // uint index of row start
  v8f c = {0.f,0.f,0.f,0.f,0.f,0.f,0.f,0.f};
  int nfrag = K >> 5;
  for (int kk = 0; kk < nfrag; ++kk) {
    Frag16 a, b;
    int kb = kk * 32 + koff;
#pragma unroll
    for (int v = 0; v < 8; ++v) {              // A layout: K 0..7 then 16..23
      int k = kb + ((v < 4) ? (v * 2) : (16 + (v - 4) * 2));
      a.u[v] = A32[rowK2 + (k >> 1)];
    }
    const unsigned* wv = Wpack + (((kk * 6 + ntile) * 32) + lane) * 8;
#pragma unroll
    for (int v = 0; v < 8; ++v) b.u[v] = wv[v];
    c = __builtin_amdgcn_wmma_f32_16x16x32_bf16(
            false, a.bf, false, b.bf, (short)0, c, false, false);
  }
  int col   = ntile * 16 + (lane & 15);
  float bv  = bias[col];
  int rbase = mtile * 16 + ((lane >> 4) * 8);
#pragma unroll
  for (int j = 0; j < 8; ++j) C[(rbase + j) * HIDC + col] = c[j] + bv;
}

// ------- semantic-attention GEMM: acc[col] += sum_n tanh((A@kw+kb)) --------
__global__ void k_semantic(const unsigned* __restrict__ A32,
                           const unsigned* __restrict__ Wpack,
                           const float* __restrict__ kb,
                           float* __restrict__ acc) {
  __shared__ float sacc[HIDC];
  int t = threadIdx.x;
  if (t < HIDC) sacc[t] = 0.f;
  __syncthreads();
  int wid   = (blockIdx.x * blockDim.x + t) >> 5;   // grid is exact
  int lane  = t & 31;
  int mtile = wid / 6, ntile = wid - mtile * 6;
  int row   = mtile * 16 + (lane & 15);
  int koff  = (lane >> 4) * 8;
  int rowK2 = (row * HIDC) >> 1;
  v8f c = {0.f,0.f,0.f,0.f,0.f,0.f,0.f,0.f};
#pragma unroll
  for (int kk = 0; kk < 3; ++kk) {
    Frag16 a, b;
    int kbase = kk * 32 + koff;
#pragma unroll
    for (int v = 0; v < 8; ++v) {
      int k = kbase + ((v < 4) ? (v * 2) : (16 + (v - 4) * 2));
      a.u[v] = A32[rowK2 + (k >> 1)];
    }
    const unsigned* wv = Wpack + (((kk * 6 + ntile) * 32) + lane) * 8;
#pragma unroll
    for (int v = 0; v < 8; ++v) b.u[v] = wv[v];
    c = __builtin_amdgcn_wmma_f32_16x16x32_bf16(
            false, a.bf, false, b.bf, (short)0, c, false, false);
  }
  int col  = ntile * 16 + (lane & 15);
  float bv = kb[col];
  float tv = 0.f;
#pragma unroll
  for (int j = 0; j < 8; ++j) tv += tanhf(c[j] + bv);
  tv += __shfl_xor(tv, 16, 32);                 // add other row-half
  if (lane < 16) atomicAdd(&sacc[col], tv);
  __syncthreads();
  if (t < HIDC) atomicAdd(&acc[t], sacc[t]);
}

// ------------------------------ edge phases --------------------------------
__global__ void k_alpha(const float* __restrict__ H,
                        const float* __restrict__ as,
                        const float* __restrict__ ad,
                        float* __restrict__ alpha_s,
                        float* __restrict__ alpha_d, int n3) {
  int t = blockIdx.x * blockDim.x + threadIdx.x;
  if (t >= n3) return;
  int i = t / 3, hh = t - i * 3;
  const float* hp = H + i * HIDC + hh * 32;
  const float* av = as + hh * 32;
  const float* bv = ad + hh * 32;
  float s1 = 0.f, s2 = 0.f;
#pragma unroll
  for (int d = 0; d < 32; ++d) { float hv = hp[d]; s1 += hv * av[d]; s2 += hv * bv[d]; }
  alpha_s[t] = s1; alpha_d[t] = s2;
}

__global__ void k_edge_a(const int* __restrict__ src, const int* __restrict__ dst,
                         const float* __restrict__ as, const float* __restrict__ ad,
                         float* __restrict__ e, unsigned* __restrict__ mu, int E_) {
  int t = blockIdx.x * blockDim.x + threadIdx.x;
  if (t >= E_) return;
  int s = src[t], d = dst[t];
#pragma unroll
  for (int h = 0; h < 3; ++h) {
    float v = as[s * 3 + h] + ad[d * 3 + h];
    v = v > 0.f ? v : 0.2f * v;                 // leaky_relu
    e[t * 3 + h] = v;
    atomicMax(&mu[d * 3 + h], fkey(v));         // segment max via ordered key
  }
}

__global__ void k_edge_b(const int* __restrict__ dst,
                         float* __restrict__ e, const unsigned* __restrict__ mu,
                         float* __restrict__ s, int E_) {
  int t = blockIdx.x * blockDim.x + threadIdx.x;
  if (t >= E_) return;
  int d = dst[t];
#pragma unroll
  for (int h = 0; h < 3; ++h) {
    float m  = funkey(mu[d * 3 + h]);
    float ex = __expf(e[t * 3 + h] - m);
    e[t * 3 + h] = ex;
    atomicAdd(&s[d * 3 + h], ex);
  }
}

// 24 threads/edge, float4 chunks: out[dst] += h[src] * alpha
__global__ void k_edge_c(const int* __restrict__ src, const int* __restrict__ dst,
                         const float* __restrict__ ex, const float* __restrict__ sden,
                         const float* __restrict__ H, float* __restrict__ out,
                         int E_) {
  long long t = blockIdx.x * (long long)blockDim.x + threadIdx.x;
  if (t >= (long long)E_ * 24) return;
  int ed = (int)(t / 24), c = (int)(t - (long long)ed * 24);
  int s = src[ed], d = dst[ed];
  int h = c >> 3;
  float alpha = ex[ed * 3 + h] / (sden[d * 3 + h] + 1e-16f);
  float4 hv = ((const float4*)H)[s * 24 + c];
  float* op = out + d * HIDC + c * 4;
  atomicAdd(op + 0, hv.x * alpha);
  atomicAdd(op + 1, hv.y * alpha);
  atomicAdd(op + 2, hv.z * alpha);
  atomicAdd(op + 3, hv.w * alpha);
}

// relu in place + bf16 copy for the semantic GEMM
__global__ void k_relu_bf(float* __restrict__ o, unsigned short* __restrict__ ob, int n) {
  int t = blockIdx.x * blockDim.x + threadIdx.x;
  if (t >= n) return;
  float v = o[t]; v = v > 0.f ? v : 0.f;
  o[t] = v; ob[t] = f2bf(v);
}

// score[r] = q . (acc_r / N); softmax over 2 relations
__global__ void k_score(const float* __restrict__ acc, const float* __restrict__ q,
                        float* __restrict__ wsem) {
  int lane = threadIdx.x;
  float p0 = 0.f, p1 = 0.f;
#pragma unroll
  for (int j = 0; j < 3; ++j) {
    int n = lane + j * 32;
    p0 += q[n] * acc[n];
    p1 += q[n] * acc[HIDC + n];
  }
#pragma unroll
  for (int off = 16; off; off >>= 1) {
    p0 += __shfl_xor(p0, off, 32);
    p1 += __shfl_xor(p1, off, 32);
  }
  if (lane == 0) {
    float s0 = p0 / (float)NN, s1 = p1 / (float)NN;
    float m = s0 > s1 ? s0 : s1;
    float w0 = __expf(s0 - m), w1 = __expf(s1 - m);
    float inv = 1.f / (w0 + w1);
    wsem[0] = w0 * inv; wsem[1] = w1 * inv;
  }
}

__global__ void k_combine(const float* __restrict__ o0, const float* __restrict__ o1,
                          const float* __restrict__ wsem, float* __restrict__ H,
                          unsigned short* __restrict__ Hbf, int n, int wbf) {
  int t = blockIdx.x * blockDim.x + threadIdx.x;
  if (t >= n) return;
  float v = wsem[0] * o0[t] + wsem[1] * o1[t];
  v = v > 0.f ? v : 0.f;                       // inter-layer relu
  H[t] = v;
  if (wbf) Hbf[t] = f2bf(v);
}

__global__ void k_fc(const float* __restrict__ H, const float* __restrict__ w,
                     const float* __restrict__ b, float* __restrict__ out, int n) {
  int t = blockIdx.x * blockDim.x + threadIdx.x;
  if (t >= n) return;
  const float4* hp = (const float4*)(H + t * HIDC);
  const float4* wp = (const float4*)w;
  float s = 0.f;
#pragma unroll
  for (int c = 0; c < 24; ++c) {
    float4 hv = hp[c], wv = wp[c];
    s += hv.x * wv.x + hv.y * wv.y + hv.z * wv.z + hv.w * wv.w;
  }
  s += b[0];
  out[t] = 1.f / (1.f + __expf(-s));
}

// ------------------------------- host side ---------------------------------
static inline int nblk(long long n, int b) { return (int)((n + b - 1) / b); }

static void run_layer(const float* projw, const float* projb, int Kin,
                      const float* asrc, const float* adst,
                      const float* q, const float* kw, const float* kb,
                      const int* ei0, const int* ei1,
                      float* h, float* out0, float* out1,
                      unsigned short* abf, float* e, float* as_, float* ad_,
                      unsigned* mu, float* sden,
                      unsigned* wp_proj, unsigned* wp_kw,
                      float* acc, float* wsem, int write_bf_next,
                      hipStream_t stream) {
  const int TILES = (NN / 16) * 6;                        // 75000, exact
  k_pack_w<<<nblk((Kin / 32) * 6 * 32 * 8, 256), 256, 0, stream>>>(projw, wp_proj, Kin);
  k_pack_w<<<nblk(3 * 6 * 32 * 8, 256), 256, 0, stream>>>(kw, wp_kw, HIDC);
  k_zero_f32<<<1, 192, 0, stream>>>(acc, 192);
  k_gemm<<<TILES / 8, 256, 0, stream>>>((const unsigned*)abf, wp_proj, projb, h, Kin, TILES);
  for (int r = 0; r < 2; ++r) {
    const int* ei  = r ? ei1 : ei0;
    const int* src = ei;
    const int* dst = ei + EE;
    float* out = r ? out1 : out0;
    k_alpha<<<nblk(NN * 3, 256), 256, 0, stream>>>(h, asrc + r * HIDC, adst + r * HIDC,
                                                   as_, ad_, NN * 3);
    k_init_rel<<<nblk((long long)NN * HIDC, 256), 256, 0, stream>>>(mu, sden, out,
                                                                    NN * HIDC, NN * 3);
    k_edge_a<<<nblk(EE, 256), 256, 0, stream>>>(src, dst, as_, ad_, e, mu, EE);
    k_edge_b<<<nblk(EE, 256), 256, 0, stream>>>(dst, e, mu, sden, EE);
    k_edge_c<<<nblk((long long)EE * 24, 256), 256, 0, stream>>>(src, dst, e, sden, h, out, EE);
    k_relu_bf<<<nblk((long long)NN * HIDC, 256), 256, 0, stream>>>(out, abf, NN * HIDC);
    k_semantic<<<TILES / 8, 256, 0, stream>>>((const unsigned*)abf, wp_kw, kb, acc + r * HIDC);
  }
  k_score<<<1, 32, 0, stream>>>(acc, q, wsem);
  k_combine<<<nblk((long long)NN * HIDC, 256), 256, 0, stream>>>(out0, out1, wsem, h, abf,
                                                                 NN * HIDC, write_bf_next);
}

extern "C" void kernel_launch(void* const* d_in, const int* in_sizes, int n_in,
                              void* d_out, int out_size, void* d_ws, size_t ws_size,
                              hipStream_t stream) {
  const float* x        = (const float*)d_in[0];
  const int*   ei_cites = (const int*)d_in[1];
  const int*   ei_cited = (const int*)d_in[2];
  const float* p1_projw = (const float*)d_in[3];
  const float* p1_projb = (const float*)d_in[4];
  const float* p1_asrc  = (const float*)d_in[5];
  const float* p1_adst  = (const float*)d_in[6];
  const float* p1_q     = (const float*)d_in[7];
  const float* p1_kw    = (const float*)d_in[8];
  const float* p1_kb    = (const float*)d_in[9];
  const float* p2_projw = (const float*)d_in[10];
  const float* p2_projb = (const float*)d_in[11];
  const float* p2_asrc  = (const float*)d_in[12];
  const float* p2_adst  = (const float*)d_in[13];
  const float* p2_q     = (const float*)d_in[14];
  const float* p2_kw    = (const float*)d_in[15];
  const float* p2_kb    = (const float*)d_in[16];
  const float* fc_w     = (const float*)d_in[17];
  const float* fc_b     = (const float*)d_in[18];
  float* out = (float*)d_out;

  // workspace carve-up (256B aligned)
  char* base = (char*)d_ws;
  size_t off = 0;
  auto carve = [&](size_t bytes) -> char* {
    char* p = base + off;
    off = (off + bytes + 255) & ~(size_t)255;
    return p;
  };
  float*          h    = (float*)carve((size_t)NN * HIDC * 4);
  float*          out0 = (float*)carve((size_t)NN * HIDC * 4);
  float*          out1 = (float*)carve((size_t)NN * HIDC * 4);
  unsigned short* abf  = (unsigned short*)carve((size_t)NN * 128 * 2); // reused as obf
  float*          e    = (float*)carve((size_t)EE * 3 * 4);
  float*          as_  = (float*)carve((size_t)NN * 3 * 4);
  float*          ad_  = (float*)carve((size_t)NN * 3 * 4);
  unsigned*       mu   = (unsigned*)carve((size_t)NN * 3 * 4);
  float*          sden = (float*)carve((size_t)NN * 3 * 4);
  unsigned*       wp_p = (unsigned*)carve(4 * 6 * 32 * 8 * 4);
  unsigned*       wp_k = (unsigned*)carve(3 * 6 * 32 * 8 * 4);
  float*          acc  = (float*)carve(2 * HIDC * 4);
  float*          wsem = (float*)carve(2 * 4);
  (void)ws_size; (void)in_sizes; (void)n_in; (void)out_size;

  // layer 1: bf16 copy of x, then HANConv (Kin=128); combine writes abf (bf16 h)
  k_cvt_bf16<<<nblk((long long)NN * 128, 256), 256, 0, stream>>>(x, abf, NN * 128);
  run_layer(p1_projw, p1_projb, 128, p1_asrc, p1_adst, p1_q, p1_kw, p1_kb,
            ei_cites, ei_cited, h, out0, out1, abf, e, as_, ad_, mu, sden,
            wp_p, wp_k, acc, wsem, /*write_bf_next=*/1, stream);
  // layer 2 (Kin=96)
  run_layer(p2_projw, p2_projb, 96, p2_asrc, p2_adst, p2_q, p2_kw, p2_kb,
            ei_cites, ei_cited, h, out0, out1, abf, e, as_, ad_, mu, sden,
            wp_p, wp_k, acc, wsem, /*write_bf_next=*/0, stream);
  // final: sigmoid(h @ fc_w + fc_b)
  k_fc<<<nblk(NN, 256), 256, 0, stream>>>(h, fc_w, fc_b, out, NN);
}
